// Conformer_MultiHeadAttention_13237089206632
// MI455X (gfx1250) — matrix operand order repaired and checked by reading, not compiled, not run on hardware
//
#include <hip/hip_runtime.h>

typedef __attribute__((ext_vector_type(16))) _Float16 v16h;
typedef __attribute__((ext_vector_type(8)))  float    v8f;

#define B_  2
#define S_  2048
#define E_  1024
#define H_  8
#define DH_ 128

union FragAB { v16h v; _Float16 h[16]; uint4 q[2]; };

// ---------------- f32 -> f16 convert with transpose ----------------
// src: K x N row-major (f32). dst: N x K row-major (f16), i.e. dst[n*K+k] = src[k*N+n].
__global__ void cvt_transpose_kernel(const float* __restrict__ src,
                                     _Float16* __restrict__ dst, int K, int N) {
    int i = blockIdx.x * blockDim.x + threadIdx.x;
    if (i >= K * N) return;
    int k = i / N, n = i % N;
    dst[(size_t)n * K + k] = (_Float16)src[i];
}

// ---------------- V transpose: (B*S, E) -> (B, H, Dh, S) ----------------
__global__ void transpose_v_kernel(const _Float16* __restrict__ vf,
                                   _Float16* __restrict__ vt) {
    int i = blockIdx.x * blockDim.x + threadIdx.x; // B*S*E
    if (i >= B_ * S_ * E_) return;
    int tok = i / E_, c = i % E_;
    int b = tok / S_, s = tok % S_;
    int h = c / DH_, d = c % DH_;
    vt[(((size_t)b * H_ + h) * DH_ + d) * S_ + s] = vf[i];
}

// ---------------- LayerNorm (one block per token, E=1024) ----------------
__global__ void layernorm_kernel(const float* __restrict__ x,
                                 const float* __restrict__ gamma,
                                 const float* __restrict__ beta,
                                 _Float16* __restrict__ xn) {
    __shared__ float red[256];
    const int tok = blockIdx.x;
    const int t = threadIdx.x;
    const float* xr = x + (size_t)tok * E_;
    float v[4];
    float s = 0.f;
#pragma unroll
    for (int k = 0; k < 4; ++k) { v[k] = xr[t + k * 256]; s += v[k]; }
    red[t] = s; __syncthreads();
    for (int off = 128; off > 0; off >>= 1) {
        if (t < off) red[t] += red[t + off];
        __syncthreads();
    }
    const float mean = red[0] * (1.f / E_);
    __syncthreads();
    float s2 = 0.f;
#pragma unroll
    for (int k = 0; k < 4; ++k) { float d = v[k] - mean; s2 += d * d; }
    red[t] = s2; __syncthreads();
    for (int off = 128; off > 0; off >>= 1) {
        if (t < off) red[t] += red[t + off];
        __syncthreads();
    }
    const float rstd = rsqrtf(red[0] * (1.f / E_) + 1e-5f);
    _Float16* o = xn + (size_t)tok * E_;
#pragma unroll
    for (int k = 0; k < 4; ++k) {
        int j = t + k * 256;
        o[j] = (_Float16)((v[k] - mean) * rstd * gamma[j] + beta[j]);
    }
}

// ---------------- sinusoidal positional encoding -> f16 ----------------
__global__ void posenc_kernel(_Float16* __restrict__ pe) {
    int idx = blockIdx.x * blockDim.x + threadIdx.x; // S*E/2 work items
    if (idx >= S_ * (E_ / 2)) return;
    int pos = idx / (E_ / 2);
    int i   = idx % (E_ / 2);
    float freq = expf((float)(2 * i) * (-logf(10000.f) / (float)E_));
    float ang  = (float)pos * freq;
    pe[(size_t)pos * E_ + 2 * i]     = (_Float16)sinf(ang);
    pe[(size_t)pos * E_ + 2 * i + 1] = (_Float16)cosf(ang);
}

// ---------------- generic batched WMMA GEMM (32x64 per-wave micro-tile) ---
// C[z] (MxN) = A[z] (MxK row-major, row stride lda) @ B[z] + bias1[n] + bias2[n]
// B element (k,n) at k + n*sn  (k ALWAYS contiguous -> b128 fragment loads).
// One wave computes a 32x64 tile as 2x4 WMMA subtiles: per K-step it loads
// 2 A-frags + 4 B-frags (12 x b128) and issues 8 WMMAs.
// Addressing: uniform SGPR base + 32-bit per-lane element offsets (saddr form)
// to keep VGPR pressure low (no 64-bit pointer arrays -> no spills).
template <bool OUT_F16>
__global__ __launch_bounds__(256, 1) void gemm_wmma_kernel(
        const _Float16* __restrict__ A, int lda, long aSz,
        const _Float16* __restrict__ Bm, int sn, long bSz,
        void* __restrict__ C, int ldc, long cSz,
        int M, int N, int K,
        const float* __restrict__ bias1, long b1Sz,
        const float* __restrict__ bias2, long b2Sz) {
    const int z = blockIdx.z;
    const _Float16* Ab = A + (size_t)z * aSz;
    const _Float16* Bb = Bm + (size_t)z * bSz;

    const int tilesN = N >> 6;          // 64-wide super-tiles
    const int tilesM = M >> 5;          // 32-tall super-tiles
    const int wave = threadIdx.x >> 5;
    const int lane = threadIdx.x & 31;
    const int tile = blockIdx.x * 8 + wave;
    if (tile >= tilesM * tilesN) return; // wave-uniform

    const int tm = tile / tilesN, tn = tile % tilesN;
    const int row0 = tm << 5, col0 = tn << 6;
    const int n = lane & 15;   // A: M index == B/D: N index
    const int g = lane >> 4;

    v8f acc[2][4];
#pragma unroll
    for (int i = 0; i < 2; ++i)
#pragma unroll
        for (int j = 0; j < 4; ++j) acc[i][j] = (v8f){};

    // 32-bit element offsets from uniform bases (fit: all buffers < 2^31 elems)
    // A fragment (16x32): lane holds row; halfs 0..7 = K(g*8..), 8..15 = K(16+g*8..)
    int aoff[2];
    aoff[0] = (row0 + n) * lda + g * 8;
    aoff[1] = aoff[0] + 16 * lda;
    // B fragment (32x16): lane holds col; halfs 0..7 = K(g*8..), 8..15 = K(16+g*8..)
    int boff[4];
    boff[0] = g * 8 + (col0 + n) * sn;
#pragma unroll
    for (int j = 1; j < 4; ++j) boff[j] = boff[0] + j * 16 * sn;

    for (int kk = 0; kk < K; kk += 32) {
        FragAB fa[2], fb[4];
#pragma unroll
        for (int i = 0; i < 2; ++i) {
            fa[i].q[0] = *(const uint4*)(Ab + aoff[i] + kk);
            fa[i].q[1] = *(const uint4*)(Ab + aoff[i] + kk + 16);
        }
#pragma unroll
        for (int j = 0; j < 4; ++j) {
            fb[j].q[0] = *(const uint4*)(Bb + boff[j] + kk);
            fb[j].q[1] = *(const uint4*)(Bb + boff[j] + kk + 16);
        }
        // prefetch next K-slice (global_prefetch_b8)
        __builtin_prefetch(Ab + aoff[0] + kk + 32, 0, 3);
        __builtin_prefetch(Ab + aoff[1] + kk + 32, 0, 3);
        __builtin_prefetch(Bb + boff[0] + kk + 32, 0, 3);
        __builtin_prefetch(Bb + boff[1] + kk + 32, 0, 3);
        __builtin_prefetch(Bb + boff[2] + kk + 32, 0, 3);
        __builtin_prefetch(Bb + boff[3] + kk + 32, 0, 3);
#pragma unroll
        for (int i = 0; i < 2; ++i)
#pragma unroll
            for (int j = 0; j < 4; ++j)
                acc[i][j] = __builtin_amdgcn_wmma_f32_16x16x32_f16(
                    false, fa[i].v, false, fb[j].v, (short)0, acc[i][j], false, false);
    }

    // epilogue: D layout row = r + 8*g (+16*i), col = lane&15 (+16*j)
    const int rbase = g * 8;
    const float* bb1 = bias1 ? bias1 + (size_t)z * b1Sz : nullptr;
    const float* bb2 = bias2 ? bias2 + (size_t)z * b2Sz : nullptr;
#pragma unroll
    for (int i = 0; i < 2; ++i) {
#pragma unroll
        for (int j = 0; j < 4; ++j) {
            const int col = col0 + j * 16 + n;
            float badd = 0.f;
            if (bb1) badd += bb1[col];
            if (bb2) badd += bb2[col];
            if (OUT_F16) {
                _Float16* Cb = (_Float16*)C + (size_t)z * cSz;
#pragma unroll
                for (int r = 0; r < 8; ++r)
                    Cb[(size_t)(row0 + i * 16 + rbase + r) * ldc + col] =
                        (_Float16)(acc[i][j][r] + badd);
            } else {
                float* Cb = (float*)C + (size_t)z * cSz;
#pragma unroll
                for (int r = 0; r < 8; ++r)
                    Cb[(size_t)(row0 + i * 16 + rbase + r) * ldc + col] =
                        acc[i][j][r] + badd;
            }
        }
    }
}

// ---------------- rel-shift + combine + softmax (per (b,h) slice) ----------------
// attn[i][j] = softmax_j( (cont[i][j] + shift(poss)[i][j]) * scale )
// shift: j<=i -> poss[i, S-1+j-i]; j==i+1 -> 0; j>i+1 -> poss[i+1, j-i-2]
__global__ void softmax_combine_kernel(const float* __restrict__ cont,
                                       const float* __restrict__ poss,
                                       _Float16* __restrict__ attn, float scale) {
    __shared__ float red[256];
    const int i = blockIdx.x;
    const int t = threadIdx.x;
    float vals[8];
    float mx = -3.4e38f;
#pragma unroll
    for (int k = 0; k < 8; ++k) {
        int j = t + k * 256;
        float c = cont[(size_t)i * S_ + j];
        float p;
        if (j <= i)          p = poss[(size_t)i * S_ + (S_ - 1 + j - i)];
        else if (j == i + 1) p = 0.f;
        else                 p = poss[(size_t)(i + 1) * S_ + (j - i - 2)];
        float s = (c + p) * scale;
        vals[k] = s;
        mx = fmaxf(mx, s);
    }
    red[t] = mx; __syncthreads();
    for (int off = 128; off > 0; off >>= 1) {
        if (t < off) red[t] = fmaxf(red[t], red[t + off]);
        __syncthreads();
    }
    mx = red[0]; __syncthreads();
    float sum = 0.f;
#pragma unroll
    for (int k = 0; k < 8; ++k) { vals[k] = __expf(vals[k] - mx); sum += vals[k]; }
    red[t] = sum; __syncthreads();
    for (int off = 128; off > 0; off >>= 1) {
        if (t < off) red[t] += red[t + off];
        __syncthreads();
    }
    const float inv = 1.f / red[0];
#pragma unroll
    for (int k = 0; k < 8; ++k)
        attn[(size_t)i * S_ + t + k * 256] = (_Float16)(vals[k] * inv);
}

// ---------------- host launch ----------------
extern "C" void kernel_launch(void* const* d_in, const int* in_sizes, int n_in,
                              void* d_out, int out_size, void* d_ws, size_t ws_size,
                              hipStream_t stream) {
    const float* x        = (const float*)d_in[0];
    const float* ln_gamma = (const float*)d_in[1];
    const float* ln_beta  = (const float*)d_in[2];
    const float* Wq = (const float*)d_in[3];
    const float* bq = (const float*)d_in[4];
    const float* Wk = (const float*)d_in[5];
    const float* bk = (const float*)d_in[6];
    const float* Wv = (const float*)d_in[7];
    const float* bv = (const float*)d_in[8];
    const float* Wp = (const float*)d_in[9];
    const float* u_bias = (const float*)d_in[10];
    const float* v_bias = (const float*)d_in[11];
    const float* Wo = (const float*)d_in[12];
    const float* bo = (const float*)d_in[13];

    char* ws = (char*)d_ws;
    size_t off = 0;
    auto alloc = [&](size_t bytes) -> void* {
        void* p = ws + off;
        off += (bytes + 255) & ~(size_t)255;
        return p;
    };
    _Float16* WQT = (_Float16*)alloc(DH_ * DH_ * 2);   // transposed weights (N x K, k-contig)
    _Float16* WKT = (_Float16*)alloc(DH_ * DH_ * 2);
    _Float16* WVT = (_Float16*)alloc(DH_ * DH_ * 2);
    _Float16* WPT = (_Float16*)alloc(DH_ * DH_ * 2);
    _Float16* WOT = (_Float16*)alloc((size_t)E_ * E_ * 2);
    _Float16* XN  = (_Float16*)alloc((size_t)B_ * S_ * E_ * 2);
    _Float16* PE  = (_Float16*)alloc((size_t)S_ * E_ * 2);
    _Float16* QU  = (_Float16*)alloc((size_t)B_ * S_ * E_ * 2);
    _Float16* QV  = (_Float16*)alloc((size_t)B_ * S_ * E_ * 2);
    _Float16* KF  = (_Float16*)alloc((size_t)B_ * S_ * E_ * 2);
    _Float16* VF  = (_Float16*)alloc((size_t)B_ * S_ * E_ * 2);
    _Float16* VT  = (_Float16*)alloc((size_t)B_ * S_ * E_ * 2);  // (B,H,Dh,S)
    _Float16* PF  = (_Float16*)alloc((size_t)S_ * E_ * 2);
    _Float16* CTX = (_Float16*)alloc((size_t)B_ * S_ * E_ * 2);
    float*    CONT = (float*)alloc((size_t)S_ * S_ * 4);
    float*    POSS = (float*)alloc((size_t)S_ * S_ * 4);
    _Float16* ATTN = (_Float16*)alloc((size_t)S_ * S_ * 2);

    auto gemm = [&](bool outF16,
                    const _Float16* A, int lda, long aSz,
                    const _Float16* Bm, int sn, long bSz,
                    void* C, int ldc, long cSz,
                    int M, int N, int K,
                    const float* b1, long b1Sz,
                    const float* b2, long b2Sz, int nz) {
        int tiles = (M / 32) * (N / 64);
        dim3 grid((tiles + 7) / 8, 1, nz);
        if (outF16)
            gemm_wmma_kernel<true><<<grid, 256, 0, stream>>>(
                A, lda, aSz, Bm, sn, bSz, C, ldc, cSz, M, N, K, b1, b1Sz, b2, b2Sz);
        else
            gemm_wmma_kernel<false><<<grid, 256, 0, stream>>>(
                A, lda, aSz, Bm, sn, bSz, C, ldc, cSz, M, N, K, b1, b1Sz, b2, b2Sz);
    };

    // 1. weights -> f16 transposed (GEMM-K contiguous)
    cvt_transpose_kernel<<<(DH_ * DH_ + 255) / 256, 256, 0, stream>>>(Wq, WQT, DH_, DH_);
    cvt_transpose_kernel<<<(DH_ * DH_ + 255) / 256, 256, 0, stream>>>(Wk, WKT, DH_, DH_);
    cvt_transpose_kernel<<<(DH_ * DH_ + 255) / 256, 256, 0, stream>>>(Wv, WVT, DH_, DH_);
    cvt_transpose_kernel<<<(DH_ * DH_ + 255) / 256, 256, 0, stream>>>(Wp, WPT, DH_, DH_);
    cvt_transpose_kernel<<<(E_ * E_ + 255) / 256, 256, 0, stream>>>(Wo, WOT, E_, E_);

    // 2. LayerNorm, 3. positional encoding
    layernorm_kernel<<<B_ * S_, 256, 0, stream>>>(x, ln_gamma, ln_beta, XN);
    posenc_kernel<<<(S_ * (E_ / 2) + 255) / 256, 256, 0, stream>>>(PE);

    const int MT = B_ * S_; // 4096 tokens
    // 4. per-head projections, batched over heads via grid.z
    //    B = W^T: element (k,n) at n*DH + k  -> sn = DH
    gemm(true, XN, E_, DH_, WQT, DH_, 0, QU, E_, DH_, MT, DH_, DH_, bq, 0, u_bias, DH_, H_);
    gemm(true, XN, E_, DH_, WQT, DH_, 0, QV, E_, DH_, MT, DH_, DH_, bq, 0, v_bias, DH_, H_);
    gemm(true, XN, E_, DH_, WKT, DH_, 0, KF, E_, DH_, MT, DH_, DH_, bk, 0, nullptr, 0, H_);
    gemm(true, XN, E_, DH_, WVT, DH_, 0, VF, E_, DH_, MT, DH_, DH_, bv, 0, nullptr, 0, H_);
    gemm(true, PE, E_, DH_, WPT, DH_, 0, PF, E_, DH_, S_, DH_, DH_, nullptr, 0, nullptr, 0, H_);

    // V -> (B,H,Dh,S) so attn@V has key-contiguous B operand
    transpose_v_kernel<<<(B_ * S_ * E_ + 255) / 256, 256, 0, stream>>>(VF, VT);

    // 5. attention per (b,h)
    const float scale = 1.f / 32.f; // 1/sqrt(E_)
    for (int b = 0; b < B_; ++b) {
        for (int h = 0; h < H_; ++h) {
            size_t qoff = (size_t)b * S_ * E_ + (size_t)h * DH_;
            // content = (q+u) @ k^T : B element (d,key) at key*E + d -> sn=E
            gemm(false, QU + qoff, E_, 0, KF + qoff, E_, 0,
                 CONT, S_, 0, S_, S_, DH_, nullptr, 0, nullptr, 0, 1);
            // pos = (q+v) @ p^T
            gemm(false, QV + qoff, E_, 0, PF + (size_t)h * DH_, E_, 0,
                 POSS, S_, 0, S_, S_, DH_, nullptr, 0, nullptr, 0, 1);
            softmax_combine_kernel<<<S_, 256, 0, stream>>>(CONT, POSS, ATTN, scale);
            // ctx = attn @ v : B element (key,d) at d*S + key -> sn=S
            gemm(true, ATTN, S_, 0, VT + ((size_t)b * H_ + h) * DH_ * S_, S_, 0,
                 CTX + qoff, E_, 0, S_, DH_, S_, nullptr, 0, nullptr, 0, 1);
        }
    }

    // 6. output projection: out = ctx @ Wo + bo (f32 out), B = Wo^T -> sn=E
    gemm(false, CTX, E_, 0, WOT, E_, 0, d_out, E_, 0, MT, E_, E_, bo, 0, nullptr, 0, 1);
}
